// BiMultiHeadAttention_45990509805990
// MI455X (gfx1250) — matrix operand order repaired and checked by reading, not compiled, not run on hardware
//
#include <hip/hip_runtime.h>
#include <hip/hip_bf16.h>

// ---------------------------------------------------------------------------
// BiMultiHeadAttention (GLIP-style) for MI455X / gfx1250.
//  - all matmuls on v_wmma_f32_16x16x32_bf16 (wave32)
//  - GEMM tiles staged with GLOBAL_LOAD_ASYNC_TO_LDS_B128 (ASYNCcnt) and
//    double-buffered LDS: one workgroup barrier per K-chunk
//  - fp32 operands pre-converted to bf16 once (halves HBM traffic on v)
// ---------------------------------------------------------------------------

typedef __bf16 bf16;
typedef __attribute__((ext_vector_type(16))) __bf16 v16bf;
typedef __attribute__((ext_vector_type(8)))  __bf16 v8bf;
typedef __attribute__((ext_vector_type(4)))  __bf16 v4bf;
typedef __attribute__((ext_vector_type(8)))  float  v8f;
typedef __attribute__((ext_vector_type(4)))  float  v4f;

namespace cfg {
constexpr int B  = 32;
constexpr int VN = 576;
constexpr int LN = 128;
constexpr int VD = 1024;
constexpr int LD = 768;
constexpr int E  = 1024;
constexpr int H  = 16;
constexpr int D  = 64;
constexpr float SCALE = 0.25f;           // H^-0.5

constexpr size_t MV = (size_t)B * VN;    // 18432 vision rows
constexpr size_t ML = (size_t)B * LN;    // 4096 language rows

// workspace layout (bytes)
constexpr size_t OFF_Q   = 0;                                      // bf16 [MV,E]
constexpr size_t OFF_K   = OFF_Q   + MV * E * 2;                   // bf16 [ML,E]
constexpr size_t OFF_VV  = OFF_K   + ML * E * 2;                   // bf16 [MV,E]
constexpr size_t OFF_LV  = OFF_VV  + MV * E * 2;                   // bf16 [ML,E]
constexpr size_t OFF_S   = OFF_LV  + ML * E * 2;                   // f32  [B*H,VN,LN]
constexpr size_t OFF_XVH = OFF_S   + (size_t)B * H * VN * LN * 4;  // bf16 [MV,E]
constexpr size_t OFF_XLH = OFF_XVH + MV * E * 2;                   // bf16 [ML,E]
constexpr size_t OFF_CM  = OFF_XLH + ML * E * 2;                   // f32  [B*H,LN]
constexpr size_t OFF_CI  = OFF_CM  + (size_t)B * H * LN * 4;       // f32  [B*H,LN]
// bf16 copies of fp32 operands
constexpr size_t OFF_VB  = OFF_CI  + (size_t)B * H * LN * 4;       // bf16 [MV,VD]
constexpr size_t OFF_LB  = OFF_VB  + MV * VD * 2;                  // bf16 [ML,LD]
constexpr size_t OFF_WVQ = OFF_LB  + ML * LD * 2;                  // bf16 [E,VD]
constexpr size_t OFF_WLK = OFF_WVQ + (size_t)E * VD * 2;           // bf16 [E,LD]
constexpr size_t OFF_WVV = OFF_WLK + (size_t)E * LD * 2;           // bf16 [E,VD]
constexpr size_t OFF_WLV = OFF_WVV + (size_t)E * VD * 2;           // bf16 [E,LD]
constexpr size_t OFF_WVO = OFF_WLV + (size_t)E * LD * 2;           // bf16 [VD,E]
constexpr size_t OFF_WLO = OFF_WVO + (size_t)VD * E * 2;           // bf16 [LD,E]
} // namespace cfg

// ---------------------------------------------------------------------------
// WMMA + async helpers (CDNA5 ISA, CDNA5 ISA)
// ---------------------------------------------------------------------------

__device__ __forceinline__ v8f vzero8() {
  v8f z;
#pragma unroll
  for (int i = 0; i < 8; ++i) z[i] = 0.0f;
  return z;
}

// A fragment: 16x32 bf16, lane = row M (lane%16), khalf = lane/16.
__device__ __forceinline__ v16bf frag_a(const bf16* p, int stride, int lane) {
  const int m  = lane & 15;
  const int kh = (lane >> 4) & 1;
  const bf16* q = p + m * stride + kh * 8;
  v8bf lo = *(const v8bf*)(q);
  v8bf hi = *(const v8bf*)(q + 16);
  return __builtin_shufflevector(lo, hi, 0, 1, 2, 3, 4, 5, 6, 7,
                                 8, 9, 10, 11, 12, 13, 14, 15);
}

// B fragment: 32x16 bf16, lane = col N (lane%16), khalf = lane/16 (same K order as A).
__device__ __forceinline__ v16bf frag_b(const bf16* p, int stride, int lane) {
  const int n  = lane & 15;
  const int kh = (lane >> 4) & 1;
  const bf16* q = p + n * stride + kh * 8;
  v8bf lo = *(const v8bf*)(q);
  v8bf hi = *(const v8bf*)(q + 16);
  return __builtin_shufflevector(lo, hi, 0, 1, 2, 3, 4, 5, 6, 7,
                                 8, 9, 10, 11, 12, 13, 14, 15);
}

__device__ __forceinline__ v8f wmma_bf16(v16bf a, v16bf b, v8f c) {
  return __builtin_amdgcn_wmma_f32_16x16x32_bf16(
      false, a, false, b, (short)0, c, false, false);
}

// LDS byte-offset of a generic pointer into __shared__ (aperture low bits)
__device__ __forceinline__ unsigned lds_off(const void* p) {
  return (unsigned)(uintptr_t)p;
}

// 16-byte async copy global -> LDS (GLOBAL_LOAD_ASYNC_TO_LDS_B128, ASYNCcnt)
__device__ __forceinline__ void async_copy16(unsigned lds, const void* gptr) {
  asm volatile("global_load_async_to_lds_b128 %0, %1, off"
               :: "v"(lds), "v"(gptr) : "memory");
}
__device__ __forceinline__ void wait_async0() {
  asm volatile("s_wait_asynccnt 0" ::: "memory");
}

__device__ __forceinline__ void storeO(float* p, float v) { *p = v; }
__device__ __forceinline__ void storeO(bf16* p, float v)  { *p = (bf16)v; }

// ---------------------------------------------------------------------------
// Kernel 0: fp32 -> bf16 elementwise (n % 4 == 0)
// ---------------------------------------------------------------------------
__global__ void __launch_bounds__(256)
f32_to_bf16(const float* __restrict__ src, bf16* __restrict__ dst, int n) {
  int i = (blockIdx.x * 256 + threadIdx.x) * 4;
  if (i >= n) return;
  v4f t = *(const v4f*)&src[i];
  v4bf o;
#pragma unroll
  for (int j = 0; j < 4; ++j) o[j] = (bf16)t[j];
  *(v4bf*)&dst[i] = o;
}

// ---------------------------------------------------------------------------
// Kernel 1/5: C[M,N] = alpha * (A[M,K] @ W[N,K]^T + bias[N]),  A,W bf16.
// Tile 128x128, BK=32, 256 threads = 8 waves. Wave w owns M-sub w and all
// 8 N-subs (8 WMMAs per chunk). Double-buffered async LDS staging; a single
// workgroup barrier per K-chunk. Requires M%128==0, N%128==0, K%32==0.
// ---------------------------------------------------------------------------
template <typename OT>
__global__ void __launch_bounds__(256)
gemm_bf16(const bf16* __restrict__ A, const bf16* __restrict__ W,
          const float* __restrict__ bias, OT* __restrict__ C,
          int M, int N, int K, float alpha) {
  __shared__ bf16 As[2][128 * 32];   // 2 x 8 KB
  __shared__ bf16 Ws[2][128 * 32];   // 2 x 8 KB

  const int tid  = threadIdx.x;
  const int lane = tid & 31;
  const int wv   = tid >> 5;
  const int rowBase = blockIdx.x * 128;
  const int colBase = blockIdx.y * 128;

  // staging: 128 rows x 64 B = 512 16B-slots per tile; 2 slots per thread
  auto stage = [&](int buf, int k0) {
#pragma unroll
    for (int j = 0; j < 2; ++j) {
      int s   = tid + 256 * j;           // 0..511
      int r   = s >> 2;                  // 0..127
      int seg = (s & 3) * 8;             // bf16 elements (16 B)
      async_copy16(lds_off(&As[buf][r * 32 + seg]),
                   &A[(size_t)(rowBase + r) * K + k0 + seg]);
      async_copy16(lds_off(&Ws[buf][r * 32 + seg]),
                   &W[(size_t)(colBase + r) * K + k0 + seg]);
    }
  };

  v8f acc[8];
#pragma unroll
  for (int n = 0; n < 8; ++n) acc[n] = vzero8();

  stage(0, 0);
  int buf = 0;
  for (int k0 = 0; k0 < K; k0 += 32, buf ^= 1) {
    wait_async0();          // this wave's async stores to LDS landed
    __syncthreads();        // everyone's landed; prior reads of buf^1 done
    if (k0 + 32 < K) stage(buf ^ 1, k0 + 32);

    v16bf af = frag_a(&As[buf][(wv * 16) * 32], 32, lane);
#pragma unroll
    for (int n = 0; n < 8; ++n) {
      v16bf bfr = frag_b(&Ws[buf][(n * 16) * 32], 32, lane);
      acc[n] = wmma_bf16(af, bfr, acc[n]);
    }
  }

  const int nlo = lane & 15;
  const int mb  = (lane >> 4) * 8;
#pragma unroll
  for (int n = 0; n < 8; ++n) {
    int col  = colBase + n * 16 + nlo;
    float bv = bias[col];
#pragma unroll
    for (int r = 0; r < 8; ++r) {
      float val  = alpha * (acc[n][r] + bv);
      size_t row = (size_t)rowBase + wv * 16 + mb + r;
      storeO(&C[row * (size_t)N + col], val);
    }
  }
}

// ---------------------------------------------------------------------------
// Kernel 2: logits S[bh,q,k] = q_head . k_head^T (scale folded into q).
// 64(q) x 64(k) tile per block, K-dim = D = 64, async LDS staging.
// ---------------------------------------------------------------------------
__global__ void __launch_bounds__(256)
attn_logits(const bf16* __restrict__ Q, const bf16* __restrict__ Kb,
            float* __restrict__ S) {
  using namespace cfg;
  __shared__ bf16 Asf[64 * 32];
  __shared__ bf16 Bsf[64 * 32];

  const int tid  = threadIdx.x;
  const int lane = tid & 31;
  const int wv   = tid >> 5;
  const int qt = blockIdx.x;             // 0..8  (VN/64)
  const int nt = blockIdx.y;             // 0..1  (LN/64)
  const int bh = blockIdx.z;             // 0..511
  const int b = bh >> 4, h = bh & 15;

  const bf16* Ah = Q  + (size_t)b * VN * E + h * D;   // [VN,D] row-stride E
  const bf16* Bh = Kb + (size_t)b * LN * E + h * D;   // [LN,D] row-stride E
  float* Sp = S + (size_t)bh * VN * LN;

  const int msub  = wv & 3;
  const int nbase = (wv >> 2) * 2;
  const int srw  = tid >> 2;             // 0..63 staging row
  const int sseg = (tid & 3) * 8;        // staging 16B segment (bf16 elems)

  v8f acc[2];
#pragma unroll
  for (int i = 0; i < 2; ++i) acc[i] = vzero8();

  for (int k0 = 0; k0 < D; k0 += 32) {
    async_copy16(lds_off(&Asf[srw * 32 + sseg]),
                 &Ah[(size_t)(qt * 64 + srw) * E + k0 + sseg]);
    async_copy16(lds_off(&Bsf[srw * 32 + sseg]),
                 &Bh[(size_t)(nt * 64 + srw) * E + k0 + sseg]);
    wait_async0();
    __syncthreads();

    v16bf af = frag_a(&Asf[(msub * 16) * 32], 32, lane);
#pragma unroll
    for (int i = 0; i < 2; ++i) {
      v16bf bfr = frag_b(&Bsf[((nbase + i) * 16) * 32], 32, lane);
      acc[i] = wmma_bf16(af, bfr, acc[i]);
    }
    __syncthreads();
  }

  const int nlo = lane & 15;
  const int mb  = (lane >> 4) * 8;
#pragma unroll
  for (int i = 0; i < 2; ++i) {
#pragma unroll
    for (int r = 0; r < 8; ++r) {
      int row = qt * 64 + msub * 16 + mb + r;
      int col = nt * 64 + (nbase + i) * 16 + nlo;
      Sp[(size_t)row * LN + col] = acc[i][r];
    }
  }
}

// ---------------------------------------------------------------------------
// Kernel 3: attn_v path. Row softmax over k (LN=128, + mask), then
// xv_head[64q x 64d] = P @ lv_head. One block per (b,h,q-tile of 64).
// ---------------------------------------------------------------------------
__global__ void __launch_bounds__(256)
softmax_pv(const float* __restrict__ S, const float* __restrict__ mask,
           const bf16* __restrict__ LV, bf16* __restrict__ XVH) {
  using namespace cfg;
  __shared__ float mS[LN];
  __shared__ bf16  Pb[64 * LN];          // P, row-major [q][k]   (16 KB)
  __shared__ bf16  Bt[64 * LN];          // lv^T,        [d][k]   (16 KB)

  const int tid  = threadIdx.x;
  const int lane = tid & 31;
  const int wv   = tid >> 5;
  const int qt = blockIdx.x;             // 0..8
  const int bh = blockIdx.y;             // 0..511
  const int b = bh >> 4, h = bh & 15;

  if (tid < LN) mS[tid] = mask[tid];
  __syncthreads();

  // row softmax: 4 threads per row, 32 logits each
  const int r = tid >> 2;
  const int p = tid & 3;
  const float* srow =
      S + (size_t)bh * VN * LN + (size_t)(qt * 64 + r) * LN + p * 32;

  float x[32];
  float m = -3.4e38f;
#pragma unroll
  for (int i = 0; i < 32; ++i) {
    x[i] = srow[i] + mS[p * 32 + i];
    m = fmaxf(m, x[i]);
  }
  m = fmaxf(m, __shfl_xor(m, 1));
  m = fmaxf(m, __shfl_xor(m, 2));
  float s = 0.0f;
#pragma unroll
  for (int i = 0; i < 32; ++i) {
    x[i] = __expf(x[i] - m);
    s += x[i];
  }
  s += __shfl_xor(s, 1);
  s += __shfl_xor(s, 2);
  const float inv = 1.0f / s;
#pragma unroll
  for (int i = 0; i < 32; ++i) Pb[r * LN + p * 32 + i] = (bf16)(x[i] * inv);

  // stage lv^T: Bt[d][k] = lv[b,k,h,d]
  const bf16* lvh = LV + (size_t)b * LN * E + h * D;
#pragma unroll
  for (int j = 0; j < 32; ++j) {
    int g  = tid + 256 * j;              // 0..8191
    int kk = g >> 6;                     // 0..127
    int d  = g & 63;
    Bt[d * LN + kk] = lvh[(size_t)kk * E + d];
  }
  __syncthreads();

  const int msub  = wv & 3;
  const int nbase = (wv >> 2) * 2;
  v8f acc[2];
#pragma unroll
  for (int i = 0; i < 2; ++i) acc[i] = vzero8();

#pragma unroll
  for (int k0 = 0; k0 < LN; k0 += 32) {
    v16bf af = frag_a(&Pb[(msub * 16) * LN + k0], LN, lane);
#pragma unroll
    for (int i = 0; i < 2; ++i) {
      v16bf bfr = frag_b(&Bt[((nbase + i) * 16) * LN + k0], LN, lane);
      acc[i] = wmma_bf16(af, bfr, acc[i]);
    }
  }

  const int nlo = lane & 15;
  const int mb  = (lane >> 4) * 8;
  bf16* out = XVH + ((size_t)b * VN + qt * 64) * E + h * D;
#pragma unroll
  for (int i = 0; i < 2; ++i) {
#pragma unroll
    for (int rr = 0; rr < 8; ++rr) {
      out[(size_t)(msub * 16 + mb + rr) * E + (nbase + i) * 16 + nlo] =
          (bf16)acc[i][rr];
    }
  }
}

// ---------------------------------------------------------------------------
// Kernel 4a: column stats for transposed softmax (over q, VN=576).
// ---------------------------------------------------------------------------
__global__ void __launch_bounds__(128)
col_stats(const float* __restrict__ S, float* __restrict__ CM,
          float* __restrict__ CI) {
  using namespace cfg;
  const int bh = blockIdx.x;
  const int k  = threadIdx.x;
  const float* Sp = S + (size_t)bh * VN * LN + k;
  float m = -3.4e38f, s = 0.0f;
  for (int q = 0; q < VN; ++q) {
    float x  = Sp[(size_t)q * LN];
    float nm = fmaxf(m, x);
    s = s * __expf(m - nm) + __expf(x - nm);
    m = nm;
  }
  CM[bh * LN + k] = m;
  CI[bh * LN + k] = 1.0f / s;
}

// ---------------------------------------------------------------------------
// Kernel 4b: attn_l path. xl_head[128k x 64d] = P_l^T @ vv_head,
// P_l[q,k] = exp(S[q,k]-cm[k])*ci[k]. One block per (b,h); stream q in 32s.
// ---------------------------------------------------------------------------
__global__ void __launch_bounds__(256)
softmax_t_pv(const float* __restrict__ S, const float* __restrict__ CM,
             const float* __restrict__ CI, const bf16* __restrict__ VV,
             bf16* __restrict__ XLH) {
  using namespace cfg;
  __shared__ bf16  As2[LN * 32];         // P^T chunk [k][q]  (8 KB)
  __shared__ bf16  Bs2[64 * 32];         // vv^T chunk [d][q] (4 KB)
  __shared__ float cm[LN], ci[LN];

  const int tid  = threadIdx.x;
  const int lane = tid & 31;
  const int wv   = tid >> 5;
  const int bh = blockIdx.x;
  const int b = bh >> 4, h = bh & 15;

  if (tid < LN) {
    cm[tid] = CM[bh * LN + tid];
    ci[tid] = CI[bh * LN + tid];
  }
  __syncthreads();

  const float* Sp  = S + (size_t)bh * VN * LN;
  const bf16*  vvh = VV + (size_t)b * VN * E + h * D;

  v8f acc[4];
#pragma unroll
  for (int n = 0; n < 4; ++n) acc[n] = vzero8();

  for (int q0 = 0; q0 < VN; q0 += 32) {
#pragma unroll
    for (int j = 0; j < 16; ++j) {
      int g  = tid + 256 * j;            // 0..4095
      int qq = g >> 7;                   // 0..31
      int kk = g & 127;
      float xv_ = Sp[(size_t)(q0 + qq) * LN + kk];
      As2[kk * 32 + qq] = (bf16)(__expf(xv_ - cm[kk]) * ci[kk]);
    }
#pragma unroll
    for (int j = 0; j < 8; ++j) {
      int g  = tid + 256 * j;            // 0..2047
      int qq = g >> 6;
      int d  = g & 63;
      Bs2[d * 32 + qq] = vvh[(size_t)(q0 + qq) * E + d];
    }
    __syncthreads();

    v16bf af = frag_a(&As2[(wv * 16) * 32], 32, lane);
#pragma unroll
    for (int n = 0; n < 4; ++n) {
      v16bf bfr = frag_b(&Bs2[(n * 16) * 32], 32, lane);
      acc[n] = wmma_bf16(af, bfr, acc[n]);
    }
    __syncthreads();
  }

  const int nlo = lane & 15;
  const int mb  = (lane >> 4) * 8;
  bf16* out = XLH + (size_t)b * LN * E + h * D;
#pragma unroll
  for (int n = 0; n < 4; ++n) {
#pragma unroll
    for (int rr = 0; rr < 8; ++rr) {
      out[(size_t)(wv * 16 + mb + rr) * E + n * 16 + nlo] = (bf16)acc[n][rr];
    }
  }
}

// ---------------------------------------------------------------------------
// Host-side launch
// ---------------------------------------------------------------------------
extern "C" void kernel_launch(void* const* d_in, const int* in_sizes, int n_in,
                              void* d_out, int out_size, void* d_ws,
                              size_t ws_size, hipStream_t stream) {
  using namespace cfg;
  (void)in_sizes; (void)n_in; (void)out_size; (void)ws_size;

  const float* v    = (const float*)d_in[0];
  const float* l    = (const float*)d_in[1];
  const float* mask = (const float*)d_in[2];
  const float* Wvq  = (const float*)d_in[3];
  const float* bvq  = (const float*)d_in[4];
  const float* Wlk  = (const float*)d_in[5];
  const float* blk  = (const float*)d_in[6];
  const float* Wvv  = (const float*)d_in[7];
  const float* bvv  = (const float*)d_in[8];
  const float* Wlv  = (const float*)d_in[9];
  const float* blv  = (const float*)d_in[10];
  const float* Wvo  = (const float*)d_in[11];
  const float* bvo  = (const float*)d_in[12];
  const float* Wlo  = (const float*)d_in[13];
  const float* blo  = (const float*)d_in[14];

  char* ws = (char*)d_ws;
  bf16*  qb   = (bf16*)(ws + OFF_Q);
  bf16*  kb   = (bf16*)(ws + OFF_K);
  bf16*  vvb  = (bf16*)(ws + OFF_VV);
  bf16*  lvb  = (bf16*)(ws + OFF_LV);
  float* Sb   = (float*)(ws + OFF_S);
  bf16*  xvh  = (bf16*)(ws + OFF_XVH);
  bf16*  xlh  = (bf16*)(ws + OFF_XLH);
  float* cmb  = (float*)(ws + OFF_CM);
  float* cib  = (float*)(ws + OFF_CI);
  bf16*  vb   = (bf16*)(ws + OFF_VB);
  bf16*  lb   = (bf16*)(ws + OFF_LB);
  bf16*  wvqb = (bf16*)(ws + OFF_WVQ);
  bf16*  wlkb = (bf16*)(ws + OFF_WLK);
  bf16*  wvvb = (bf16*)(ws + OFF_WVV);
  bf16*  wlvb = (bf16*)(ws + OFF_WLV);
  bf16*  wvob = (bf16*)(ws + OFF_WVO);
  bf16*  wlob = (bf16*)(ws + OFF_WLO);

  float* xv_out = (float*)d_out;                      // [B*VN, VD]
  float* xl_out = xv_out + MV * (size_t)VD;           // [B*LN, LD]

  const dim3 blk256(256);
  auto cvt = [&](const float* s, bf16* d, size_t n) {
    f32_to_bf16<<<dim3((unsigned)(n / 1024)), blk256, 0, stream>>>(s, d, (int)n);
  };

  // 0) one-time fp32 -> bf16 conversions
  cvt(v,   vb,   MV * VD);
  cvt(l,   lb,   ML * LD);
  cvt(Wvq, wvqb, (size_t)E * VD);
  cvt(Wlk, wlkb, (size_t)E * LD);
  cvt(Wvv, wvvb, (size_t)E * VD);
  cvt(Wlv, wlvb, (size_t)E * LD);
  cvt(Wvo, wvob, (size_t)VD * E);
  cvt(Wlo, wlob, (size_t)LD * E);

  // 1) input projections (SCALE folded into q)
  gemm_bf16<bf16><<<dim3(MV / 128, E / 128), blk256, 0, stream>>>(
      vb, wvqb, bvq, qb, (int)MV, E, VD, SCALE);
  gemm_bf16<bf16><<<dim3(ML / 128, E / 128), blk256, 0, stream>>>(
      lb, wlkb, blk, kb, (int)ML, E, LD, 1.0f);
  gemm_bf16<bf16><<<dim3(MV / 128, E / 128), blk256, 0, stream>>>(
      vb, wvvb, bvv, vvb, (int)MV, E, VD, 1.0f);
  gemm_bf16<bf16><<<dim3(ML / 128, E / 128), blk256, 0, stream>>>(
      lb, wlvb, blv, lvb, (int)ML, E, LD, 1.0f);

  // 2) attention logits S = q . k^T per (b,h)
  attn_logits<<<dim3(VN / 64, LN / 64, B * H), blk256, 0, stream>>>(qb, kb, Sb);

  // 3) attn_v: row softmax (+mask) and P @ lv
  softmax_pv<<<dim3(VN / 64, B * H), blk256, 0, stream>>>(Sb, mask, lvb, xvh);

  // 4) attn_l: column softmax stats then P_l^T @ vv
  col_stats<<<dim3(B * H), dim3(128), 0, stream>>>(Sb, cmb, cib);
  softmax_t_pv<<<dim3(B * H), blk256, 0, stream>>>(Sb, cmb, cib, vvb, xlh);

  // 5) output projections (bf16 -> fp32 into d_out)
  gemm_bf16<float><<<dim3(MV / 128, VD / 128), blk256, 0, stream>>>(
      xvh, wvob, bvo, xv_out, (int)MV, VD, E, 1.0f);
  gemm_bf16<float><<<dim3(ML / 128, LD / 128), blk256, 0, stream>>>(
      xlh, wlob, blo, xl_out, (int)ML, LD, E, 1.0f);
}
